// FFLSTM_77318001263051
// MI455X (gfx1250) — matrix operand order repaired and checked by reading, not compiled, not run on hardware
//
#include <hip/hip_runtime.h>
#include <hip/hip_bf16.h>

typedef _Float16 h16_t;
typedef __attribute__((ext_vector_type(8)))  _Float16 v8h;
typedef __attribute__((ext_vector_type(16))) _Float16 v16h;
typedef __attribute__((ext_vector_type(8)))  float    v8f;

static constexpr int SEQ   = 256;
static constexpr int BATCH = 128;
static constexpr int FEAT  = 512;
static constexpr int HID   = 1024;
static constexpr int NL    = 3;
static constexpr int G4    = 4 * HID;

// LDS chunk stride: 16 halves of fragment data + 8 halves pad = 48B.
// Keeps 16B alignment and spreads the 32-lane x 32B ds_load pattern across banks.
static constexpr int CHUNK = 24;

// ---------------- CDNA5 async global->LDS copy (ASYNCcnt-tracked DMA) ----------------
// GLOBAL_LOAD_ASYNC_TO_LDS_B128: per-lane 16B global->LDS, no VGPR round-trip.
// Low 32 bits of a flat pointer into LDS are the LDS byte offset (LDS aperture).
__device__ __forceinline__ void async_cp_b128(void* lds_dst, const void* gsrc) {
    unsigned lds_off = (unsigned)(uintptr_t)lds_dst;
    asm volatile("global_load_async_to_lds_b128 %0, %1, off"
                 :: "v"(lds_off), "v"(gsrc)
                 : "memory");
}
__device__ __forceinline__ void wait_async0() {
    asm volatile("s_wait_asynccnt 0" ::: "memory");
}

// ---------------- fragment loaders (WMMA 16x16x32 f16; contiguous k order, same as the B staging) ----------------

// A-matrix 16x32 (MxK): lane L<16 holds row m0+L, K = k0+{0..15};
// lanes 16-31 hold K = k0+16+{0..15}.  Two 16-byte chunks.
__device__ __forceinline__ v16h load_fragA(const h16_t* __restrict__ A, int m0, int ld,
                                           int k0, int lane) {
    const int m  = m0 + (lane & 15);
    const int kb = k0 + ((lane >> 4) << 4);
    const h16_t* p = A + (size_t)m * ld + kb;
    v8h lo = *(const v8h*)(p);
    v8h hi = *(const v8h*)(p + 8);
    return __builtin_shufflevector(lo, hi, 0,1,2,3,4,5,6,7,8,9,10,11,12,13,14,15);
}

__device__ __forceinline__ float sigm(float x) { return 1.0f / (1.0f + __expf(-x)); }

// ---------------- fused LSTM cell: gates GEMM + cell update (+ optional residual out) ----------
// grid = HID/16 = 64 blocks, block = 256 threads (8 waves). Wave w owns M-tile w,
// computes the 4 gate tiles {g*HID + n0 .. +16} so i/f/g/o stay in registers.
// B tiles (shared by all 8 waves) are staged through double-buffered LDS via async
// global->LDS DMA: 8x less L2 weight traffic, fill overlapped with WMMA consume.
__global__ __launch_bounds__(256) void lstm_cell_kernel(
    const h16_t* __restrict__ A, int Kin,          // input activations [BATCH, Kin] f16
    const h16_t* __restrict__ Wih,                 // [4H, Kin] f16 row-major
    const h16_t* __restrict__ Whh,                 // [4H, HID] f16 row-major
    const h16_t* __restrict__ h16_in,              // [BATCH, HID] f16 (prev h)
    const float* __restrict__ bias,                // [4H] (b_ih + b_hh)
    float*       __restrict__ c,                   // [BATCH, HID] fp32 (in/out)
    float*       __restrict__ h,                   // [BATCH, HID] fp32 (out)
    h16_t*       __restrict__ h16_out,             // [BATCH, HID] f16  (out)
    const float* __restrict__ ft_res,              // [BATCH, HID] or null: residual input
    h16_t*       __restrict__ s16_out)             // [BATCH, HID] f16: f16(h_new + ft_res)
{
    __shared__ h16_t ldsB[2][4 * 32 * CHUNK];      // 2 x 6 KB

    const int tid  = threadIdx.x;
    const int lane = tid & 31;
    const int wave = tid >> 5;
    const int m0   = wave << 4;
    const int n0   = blockIdx.x << 4;

    const int nk1 = Kin >> 5;                      // k-steps over the x-part
    const int nk  = nk1 + (HID >> 5);              // + k-steps over the h-part

    // cooperative B staging: 4 gate tiles x 32 lane-chunks x 32B, 16B async per thread
    auto fillB = [&](int buf, int s) {
        const h16_t* W  = (s < nk1) ? Wih : Whh;
        const int    ld = (s < nk1) ? Kin : HID;
        const int    k0 = ((s < nk1) ? s : (s - nk1)) << 5;
        const int chunk = tid >> 1;                // 0..127 = gate*32 + lane
        const int part  = tid & 1;
        const int g     = chunk >> 5;
        const int l     = chunk & 31;
        const int row   = g * HID + n0 + (l & 15);
        const int koff  = k0 + ((l >> 4) << 4) + (part << 3);
        async_cp_b128(&ldsB[buf][chunk * CHUNK + part * 8],
                      W + (size_t)row * ld + koff);
    };
    auto loadA = [&](int s) -> v16h {
        if (s < nk1) return load_fragA(A,      m0, Kin, s << 5,          lane);
        else         return load_fragA(h16_in, m0, HID, (s - nk1) << 5,  lane);
    };

    const v8f zero = {0.f,0.f,0.f,0.f,0.f,0.f,0.f,0.f};
    v8f acc[4] = {zero, zero, zero, zero};

    fillB(0, 0);
    v16h a_next = loadA(0);
    wait_async0();
    __syncthreads();

    for (int s = 0; s < nk; ++s) {
        v16h a = a_next;
        if (s + 1 < nk) {
            fillB((s + 1) & 1, s + 1);             // async-stage next B tile
            a_next = loadA(s + 1);                 // prefetch next A fragment
        }
        const h16_t* buf = ldsB[s & 1];
        #pragma unroll
        for (int g = 0; g < 4; ++g) {
            v16h b = *(const v16h*)(buf + (g * 32 + lane) * CHUNK);
            acc[g] = __builtin_amdgcn_wmma_f32_16x16x32_f16(
                false, a, false, b, (short)0, acc[g], false, false);
        }
        wait_async0();                             // next-tile DMA landed in LDS
        __syncthreads();
    }

    // cell update: D layout lane L, vgpr r -> m = m0 + 8*(L>>4) + r, n = n0 + (L&15)
    const int n     = n0 + (lane & 15);
    const int mbase = m0 + ((lane >> 4) << 3);
    const float bi = bias[0 * HID + n];
    const float bf = bias[1 * HID + n];
    const float bg = bias[2 * HID + n];
    const float bo = bias[3 * HID + n];
    #pragma unroll
    for (int r = 0; r < 8; ++r) {
        const size_t idx = (size_t)(mbase + r) * HID + n;
        const float gi = acc[0][r] + bi;
        const float gf = acc[1][r] + bf;
        const float gg = acc[2][r] + bg;
        const float go = acc[3][r] + bo;
        const float cn = sigm(gf) * c[idx] + sigm(gi) * tanhf(gg);
        const float hn = sigm(go) * tanhf(cn);
        c[idx] = cn;
        h[idx] = hn;
        h16_out[idx] = (h16_t)hn;
        if (ft_res)                                // fused residual: s16 = f16(h_new + ft)
            s16_out[idx] = (h16_t)(hn + ft_res[idx]);
    }
}

// ---------------- ft = A @ W_out^T + b_out (fused bias, fp32 + optional f16 out) ----------------
// grid = HID/32 = 32 blocks, block = 256 (8 waves); wave owns M-tile, 2 N-tiles.
// Same async LDS staging (2 tiles -> 64 chunks, issued by threads 0..127).
__global__ __launch_bounds__(256) void out_gemm_kernel(
    const h16_t* __restrict__ A,        // [BATCH, HID] f16
    const h16_t* __restrict__ Wout,     // [HID, HID] f16 row-major
    const float* __restrict__ b_out,    // [HID]
    float*       __restrict__ dst32,    // [BATCH, HID]
    h16_t*       __restrict__ dst16)    // [BATCH, HID] or nullptr
{
    __shared__ h16_t ldsB[2][2 * 32 * CHUNK];      // 2 x 3 KB

    const int tid  = threadIdx.x;
    const int lane = tid & 31;
    const int wave = tid >> 5;
    const int m0   = wave << 4;
    const int n0   = blockIdx.x << 5;
    const int nk   = HID >> 5;

    auto fillB = [&](int buf, int s) {
        if (tid < 128) {                            // wave-uniform predicate
            const int k0    = s << 5;
            const int chunk = tid >> 1;             // 0..63 = tile*32 + lane
            const int part  = tid & 1;
            const int t2    = chunk >> 5;
            const int l     = chunk & 31;
            const int row   = n0 + t2 * 16 + (l & 15);
            const int koff  = k0 + ((l >> 4) << 4) + (part << 3);
            async_cp_b128(&ldsB[buf][chunk * CHUNK + part * 8],
                          Wout + (size_t)row * HID + koff);
        }
    };

    const v8f zero = {0.f,0.f,0.f,0.f,0.f,0.f,0.f,0.f};
    v8f acc0 = zero, acc1 = zero;

    fillB(0, 0);
    v16h a_next = load_fragA(A, m0, HID, 0, lane);
    wait_async0();
    __syncthreads();

    for (int s = 0; s < nk; ++s) {
        v16h a = a_next;
        if (s + 1 < nk) {
            fillB((s + 1) & 1, s + 1);
            a_next = load_fragA(A, m0, HID, (s + 1) << 5, lane);
        }
        const h16_t* buf = ldsB[s & 1];
        v16h b0 = *(const v16h*)(buf + (0 * 32 + lane) * CHUNK);
        v16h b1 = *(const v16h*)(buf + (1 * 32 + lane) * CHUNK);
        acc0 = __builtin_amdgcn_wmma_f32_16x16x32_f16(false, a, false, b0, (short)0, acc0, false, false);
        acc1 = __builtin_amdgcn_wmma_f32_16x16x32_f16(false, a, false, b1, (short)0, acc1, false, false);
        wait_async0();
        __syncthreads();
    }

    const int nlo   = n0 + (lane & 15);
    const int nhi   = nlo + 16;
    const int mbase = m0 + ((lane >> 4) << 3);
    const float bl = b_out[nlo];
    const float bh = b_out[nhi];
    #pragma unroll
    for (int r = 0; r < 8; ++r) {
        const size_t row = (size_t)(mbase + r) * HID;
        const float v0 = acc0[r] + bl;
        const float v1 = acc1[r] + bh;
        dst32[row + nlo] = v0;
        dst32[row + nhi] = v1;
        if (dst16) {
            dst16[row + nlo] = (h16_t)v0;
            dst16[row + nhi] = (h16_t)v1;
        }
    }
}

// ---------------- small helper kernels ----------------
__global__ void k_f32_to_f16(const float* __restrict__ s, h16_t* __restrict__ d, int n) {
    for (int i = blockIdx.x * blockDim.x + threadIdx.x; i < n; i += gridDim.x * blockDim.x)
        d[i] = (h16_t)s[i];
}
__global__ void k_bias_sum(const float* __restrict__ a, const float* __restrict__ b,
                           float* __restrict__ o, int n) {
    for (int i = blockIdx.x * blockDim.x + threadIdx.x; i < n; i += gridDim.x * blockDim.x)
        o[i] = a[i] + b[i];
}
__global__ void k_fill_zero(unsigned int* __restrict__ p, int n) {
    for (int i = blockIdx.x * blockDim.x + threadIdx.x; i < n; i += gridDim.x * blockDim.x)
        p[i] = 0u;
}

// ---------------- host ----------------
extern "C" void kernel_launch(void* const* d_in, const int* in_sizes, int n_in,
                              void* d_out, int out_size, void* d_ws, size_t ws_size,
                              hipStream_t stream) {
    const float* x        = (const float*)d_in[0]; // [SEQ, BATCH, FEAT]
    const float* W_ih0    = (const float*)d_in[1]; // [4H, FEAT]
    const float* W_ih_r   = (const float*)d_in[2]; // [NL-1, 4H, HID]
    const float* W_hh     = (const float*)d_in[3]; // [NL, 4H, HID]
    const float* b_ih     = (const float*)d_in[4]; // [NL, 4H]
    const float* b_hh     = (const float*)d_in[5]; // [NL, 4H]
    const float* W_out    = (const float*)d_in[6]; // [HID, HID]
    const float* b_out    = (const float*)d_in[7]; // [HID]
    float* out = (float*)d_out;                    // [SEQ, BATCH, HID]

    // workspace carving (256B aligned)
    char* wp = (char*)d_ws;
    auto carve = [&](size_t bytes) -> void* {
        void* p = (void*)wp;
        wp += (bytes + 255) & ~(size_t)255;
        return p;
    };
    const size_t BH = (size_t)BATCH * HID;
    h16_t* x16      = (h16_t*)carve((size_t)SEQ * BATCH * FEAT * sizeof(h16_t));
    h16_t* Wih0_16  = (h16_t*)carve((size_t)G4 * FEAT * sizeof(h16_t));
    h16_t* Wihr16   = (h16_t*)carve((size_t)(NL - 1) * G4 * HID * sizeof(h16_t));
    h16_t* Whh16    = (h16_t*)carve((size_t)NL * G4 * HID * sizeof(h16_t));
    h16_t* Wout16   = (h16_t*)carve((size_t)HID * HID * sizeof(h16_t));
    float* bias     = (float*)carve((size_t)NL * G4 * sizeof(float));
    float* hbuf     = (float*)carve((size_t)NL * BH * sizeof(float));
    float* cbuf     = (float*)carve((size_t)NL * BH * sizeof(float));
    h16_t* h16buf   = (h16_t*)carve((size_t)2 * NL * BH * sizeof(h16_t)); // [parity][layer][BH]
    float* ft       = (float*)carve(BH * sizeof(float));
    h16_t* ft16     = (h16_t*)carve(BH * sizeof(h16_t));
    h16_t* s16      = (h16_t*)carve(BH * sizeof(h16_t));
    (void)ws_size; (void)n_in; (void)in_sizes; (void)out_size;

    // one-time prep (fully parallel, off the recurrent critical path)
    k_f32_to_f16<<<2048, 256, 0, stream>>>(x,      x16,     SEQ * BATCH * FEAT);
    k_f32_to_f16<<<512,  256, 0, stream>>>(W_ih0,  Wih0_16, G4 * FEAT);
    k_f32_to_f16<<<1024, 256, 0, stream>>>(W_ih_r, Wihr16,  (NL - 1) * G4 * HID);
    k_f32_to_f16<<<1024, 256, 0, stream>>>(W_hh,   Whh16,   NL * G4 * HID);
    k_f32_to_f16<<<256,  256, 0, stream>>>(W_out,  Wout16,  HID * HID);
    k_bias_sum  <<<48,   256, 0, stream>>>(b_ih, b_hh, bias, NL * G4);
    k_fill_zero <<<512,  256, 0, stream>>>((unsigned int*)cbuf,   (int)(NL * BH));
    k_fill_zero <<<256,  256, 0, stream>>>((unsigned int*)h16buf, (int)(NL * BH / 2)); // parity 0

    for (int t = 0; t < SEQ; ++t) {
        const h16_t* hin  = h16buf + (size_t)(t & 1) * NL * BH;
        h16_t*       hout = h16buf + (size_t)((t + 1) & 1) * NL * BH;

        // layer 0 (no residual)
        lstm_cell_kernel<<<HID / 16, 256, 0, stream>>>(
            x16 + (size_t)t * BATCH * FEAT, FEAT,
            Wih0_16, Whh16, hin, bias,
            cbuf, hbuf, hout, nullptr, nullptr);
        out_gemm_kernel<<<HID / 32, 256, 0, stream>>>(hout, Wout16, b_out, ft, ft16);

        // layer 1 (cell + fused residual s16 = f16(h1 + ft))
        lstm_cell_kernel<<<HID / 16, 256, 0, stream>>>(
            ft16, HID,
            Wihr16, Whh16 + (size_t)1 * G4 * HID, hin + 1 * BH, bias + 1 * G4,
            cbuf + 1 * BH, hbuf + 1 * BH, hout + 1 * BH, ft, s16);
        out_gemm_kernel<<<HID / 32, 256, 0, stream>>>(s16, Wout16, b_out, ft, ft16);

        // layer 2 (cell + fused residual)
        lstm_cell_kernel<<<HID / 16, 256, 0, stream>>>(
            ft16, HID,
            Wihr16 + (size_t)1 * G4 * HID, Whh16 + (size_t)2 * G4 * HID,
            hin + 2 * BH, bias + 2 * G4,
            cbuf + 2 * BH, hbuf + 2 * BH, hout + 2 * BH, ft, s16);
        out_gemm_kernel<<<HID / 32, 256, 0, stream>>>(
            s16, Wout16, b_out, out + (size_t)t * BH, nullptr);
    }
}
